// MambaEHR_90580860272641
// MI455X (gfx1250) — compile-verified, matrix operand order checked by reading, not run on hardware
//
#include <hip/hip_runtime.h>
#include <hip/hip_bf16.h>

typedef __attribute__((ext_vector_type(16))) _Float16 v16h;
typedef __attribute__((ext_vector_type(8)))  float    v8f;
typedef __attribute__((ext_vector_type(4)))  unsigned int v4u;
typedef __attribute__((ext_vector_type(8)))  int      v8i;
typedef __attribute__((ext_vector_type(4)))  int      v4i;

#define B_   32
#define L_   2048
#define TSD  128
#define SD   64
#define DM   32
#define DS   16
#define DI   64
#define DTR  2
#define DC   4

// ---- scan chunking / LDS layout (floats, per buffer) ----
#define TCH   32
#define NCH   (L_ / TCH)
#define OFF_D 0
#define OFF_X (TCH * DI)
#define OFF_Z (2 * TCH * DI)
#define OFF_B (3 * TCH * DI)
#define OFF_C (3 * TCH * DI + TCH * DS)
#define BUF_F (3 * TCH * DI + 2 * TCH * DS)   // 7168 floats = 28 KB

#if defined(__has_builtin)
#  if __has_builtin(__builtin_amdgcn_tensor_load_to_lds) && \
      __has_builtin(__builtin_amdgcn_s_wait_tensorcnt)
#    define HAVE_TDM 1
#  endif
#endif
#ifndef HAVE_TDM
#  define HAVE_TDM 0
#endif

__device__ __forceinline__ float silu_f(float v) { return v / (1.f + __expf(-v)); }

// The TDM writes LDS through a descriptor holding only an integer LDS offset,
// so `smem` would otherwise be a non-escaping, never-stored LDS global and its
// loads fold to undef.  This fence forces the pointer to escape into an asm
// with a memory clobber, so the compiler must assume the DMA wrote it.
__device__ __forceinline__ void tdm_publish_fence(void* p) {
  asm volatile("" : "+v"(p) :: "memory");
}

#if HAVE_TDM
// Issue one TDM 2D tile load: `rows` rows of `row_elems` f32, contiguous
// (tensor_dim0_stride == row_elems), into LDS at byte offset lds_byte_off.
// D# layout per CDNA5 ISA §8 (group0: count/lds/global/type, group1: dims).
__device__ __forceinline__ void tdm_load_2d(const float* g, unsigned lds_byte_off,
                                            unsigned row_elems, unsigned rows) {
  const unsigned long long ga = (unsigned long long)(const void*)g;
  v4u g0;
  g0[0] = 1u;                                         // count=1 (valid), user mode
  g0[1] = lds_byte_off;                               // lds_addr
  g0[2] = (unsigned)ga;                               // global_addr[31:0]
  g0[3] = (unsigned)((ga >> 32) & 0x01FFFFFFu)        // global_addr[56:32]
          | (2u << 30);                               // type = 2 ("image")
  v8i g1;
  g1[0] = (int)(2u << 16);                            // data_size = 4 bytes
  g1[1] = (int)((row_elems & 0xFFFFu) << 16);         // tensor_dim0[15:0]
  g1[2] = (int)(((row_elems >> 16) & 0xFFFFu)         // tensor_dim0[31:16]
          | ((rows & 0xFFFFu) << 16));                // tensor_dim1[15:0]
  g1[3] = (int)(((rows >> 16) & 0xFFFFu)              // tensor_dim1[31:16]
          | ((row_elems & 0xFFFFu) << 16));           // tile_dim0
  g1[4] = (int)(rows & 0xFFFFu);                      // tile_dim1 (tile_dim2 = 0)
  g1[5] = (int)row_elems;                             // tensor_dim0_stride[31:0]
  g1[6] = 0;
  g1[7] = 0;
  const v4i z4 = {0, 0, 0, 0};                        // groups 2/3 unused (2D)
  const v8i z8 = {0, 0, 0, 0, 0, 0, 0, 0};
  __builtin_amdgcn_tensor_load_to_lds(g0, g1, z4, z4, z8, 0);
}
#endif

// ---------------------------------------------------------------------------
// Kernel 1: masked input projection  (B*L x 128) @ (128 x 32)  via WMMA f16
// ---------------------------------------------------------------------------
__global__ __launch_bounds__(128) void k_maskproj(const float* __restrict__ tsv,
                                                  const int*   __restrict__ ind,
                                                  const float* __restrict__ Wts,
                                                  const float* __restrict__ bts,
                                                  float* __restrict__ tsp) {
  const int wave = threadIdx.x >> 5;
  const int lane = threadIdx.x & 31;
  const int tile = blockIdx.x * 4 + wave;        // 0 .. B*L/16 - 1
  const int b    = tile >> 7;                    // 128 tiles per batch
  const int row0 = (tile & 127) << 4;
  const int m    = lane & 15;
  const int half = lane >> 4;
  const int kb   = half * 8;
  const long abase = ((long)b * L_ + (row0 + m)) * TSD;

  v8f c0 = {}; v8f c1 = {};
#pragma unroll
  for (int s = 0; s < 4; ++s) {
    const int k0 = s * 32;
    v16h a;
#pragma unroll
    for (int j = 0; j < 8; ++j) {
      const int ka = k0 + kb + j;
      const int kc = k0 + 16 + kb + j;
      a[j]     = (_Float16)(tsv[abase + ka] * (float)ind[abase + ka]);
      a[8 + j] = (_Float16)(tsv[abase + kc] * (float)ind[abase + kc]);
    }
    v16h b0, b1;
#pragma unroll
    for (int j = 0; j < 8; ++j) {
      b0[j]     = (_Float16)Wts[(k0 + kb + j)      * DM + m];
      b0[8 + j] = (_Float16)Wts[(k0 + 16 + kb + j) * DM + m];
      b1[j]     = (_Float16)Wts[(k0 + kb + j)      * DM + 16 + m];
      b1[8 + j] = (_Float16)Wts[(k0 + 16 + kb + j) * DM + 16 + m];
    }
    c0 = __builtin_amdgcn_wmma_f32_16x16x32_f16(false, a, false, b0, (short)0, c0, false, false);
    c1 = __builtin_amdgcn_wmma_f32_16x16x32_f16(false, a, false, b1, (short)0, c1, false, false);
  }
  const float bias0 = bts[m], bias1 = bts[16 + m];
#pragma unroll
  for (int r = 0; r < 8; ++r) {
    const long o = ((long)b * L_ + (row0 + r + 8 * half)) * DM;
    tsp[o + m]      = c0[r] + bias0;
    tsp[o + 16 + m] = c1[r] + bias1;
  }
}

// ---------------------------------------------------------------------------
// Kernel 2: xz = ts_proj @ W_in  (B*L x 32) @ (32 x 128) via WMMA f16
// ---------------------------------------------------------------------------
__global__ __launch_bounds__(128) void k_inproj(const float* __restrict__ tsp,
                                                const float* __restrict__ Win,
                                                float* __restrict__ xr,
                                                float* __restrict__ zb) {
  const int wave = threadIdx.x >> 5;
  const int lane = threadIdx.x & 31;
  const int tile = blockIdx.x * 4 + wave;
  const int b    = tile >> 7;
  const int row0 = (tile & 127) << 4;
  const int m    = lane & 15;
  const int half = lane >> 4;
  const int kb   = half * 8;

  v16h a;
  {
    const long base = ((long)b * L_ + (row0 + m)) * DM;
#pragma unroll
    for (int j = 0; j < 8; ++j) {
      a[j]     = (_Float16)tsp[base + kb + j];
      a[8 + j] = (_Float16)tsp[base + 16 + kb + j];
    }
  }
#pragma unroll
  for (int nt = 0; nt < 8; ++nt) {
    v16h bm;
#pragma unroll
    for (int j = 0; j < 8; ++j) {
      bm[j]     = (_Float16)Win[(kb + j)      * (2 * DI) + nt * 16 + m];
      bm[8 + j] = (_Float16)Win[(16 + kb + j) * (2 * DI) + nt * 16 + m];
    }
    v8f c = {};
    c = __builtin_amdgcn_wmma_f32_16x16x32_f16(false, a, false, bm, (short)0, c, false, false);
#pragma unroll
    for (int r = 0; r < 8; ++r) {
      const int orow = row0 + r + 8 * half;
      const int col  = nt * 16 + m;
      if (col < DI) xr[((long)b * L_ + orow) * DI + col]        = c[r];
      else          zb[((long)b * L_ + orow) * DI + (col - DI)] = c[r];
    }
  }
}

// ---------------------------------------------------------------------------
// Kernel 3a: depthwise causal conv (K=4) + bias + SiLU
// ---------------------------------------------------------------------------
__global__ void k_conv(const float* __restrict__ xr, const float* __restrict__ cw,
                       const float* __restrict__ cb, float* __restrict__ xs) {
  const long idx = (long)blockIdx.x * blockDim.x + threadIdx.x;
  const int  e   = (int)(idx & 63);
  const long bt  = idx >> 6;
  const int  t   = (int)(bt & (L_ - 1));
  float acc = cb[e];
#pragma unroll
  for (int k = 0; k < DC; ++k) {
    const int tt = t - (DC - 1) + k;
    if (tt >= 0) acc += cw[e * DC + k] * xr[(bt - (DC - 1) + k) * DI + e];
  }
  xs[idx] = silu_f(acc);
}

// ---------------------------------------------------------------------------
// Kernel 3b: dbc = x @ W_x (64 -> 34); delta = softplus(dbc[:2] @ W_dt + b)
// ---------------------------------------------------------------------------
__global__ void k_dbc(const float* __restrict__ xs, const float* __restrict__ Wx,
                      const float* __restrict__ Wdt, const float* __restrict__ bdt,
                      float* __restrict__ delta, float* __restrict__ Bm,
                      float* __restrict__ Cm) {
  const long bt = (long)blockIdx.x * blockDim.x + threadIdx.x;
  float dbc[DTR + 2 * DS] = {};
  const float* xrow = xs + bt * DI;
  for (int e = 0; e < DI; ++e) {
    const float xv = xrow[e];
    const float* w = Wx + e * (DTR + 2 * DS);
#pragma unroll
    for (int j = 0; j < DTR + 2 * DS; ++j) dbc[j] += xv * w[j];
  }
#pragma unroll
  for (int j = 0; j < DS; ++j) {
    Bm[bt * DS + j] = dbc[DTR + j];
    Cm[bt * DS + j] = dbc[DTR + DS + j];
  }
  for (int e = 0; e < DI; ++e) {
    const float d = dbc[0] * Wdt[e] + dbc[1] * Wdt[DI + e] + bdt[e];
    delta[bt * DI + e] = (d > 20.f) ? d : __logf(1.f + __expf(d));
  }
}

// ---------------------------------------------------------------------------
// Kernel 4: selective scan, double-buffered TDM staging into LDS.
// One block per batch, 1024 threads = one thread per (e, n) state.
// Wave 0 lane 0 issues tensor_load_to_lds for chunk c+1 while all waves
// consume chunk c from LDS; n-reduction via 16-lane shfl_xor.
// ---------------------------------------------------------------------------
__global__ __launch_bounds__(1024) void k_scan(const float* __restrict__ xs,
                                               const float* __restrict__ delta,
                                               const float* __restrict__ Bm,
                                               const float* __restrict__ Cm,
                                               const float* __restrict__ zb,
                                               const float* __restrict__ Alog,
                                               const float* __restrict__ Dskip,
                                               float* __restrict__ y) {
  __shared__ float smem[2 * BUF_F];     // assumed at LDS offset 0 (sole shared var)
  const int b = blockIdx.x;
  const int e = threadIdx.x >> 4;
  const int n = threadIdx.x & 15;
  const float A  = -__expf(Alog[e * DS + n]);
  const float Dk = Dskip[e];
  const long base_e = (long)b * L_ * DI;
  const long base_n = (long)b * L_ * DS;
  float h = 0.f;

  // ---- stage chunk 0 into buffer 0 ----
#if HAVE_TDM
  if (threadIdx.x == 0) {
    tdm_load_2d(delta + base_e, OFF_D * 4, DI, TCH);
    tdm_load_2d(xs    + base_e, OFF_X * 4, DI, TCH);
    tdm_load_2d(zb    + base_e, OFF_Z * 4, DI, TCH);
    tdm_load_2d(Bm    + base_n, OFF_B * 4, DS, TCH);
    tdm_load_2d(Cm    + base_n, OFF_C * 4, DS, TCH);
    __builtin_amdgcn_s_wait_tensorcnt(0);
  }
#else
  for (int i = threadIdx.x; i < TCH * DI; i += 1024) {
    smem[OFF_D + i] = delta[base_e + i];
    smem[OFF_X + i] = xs[base_e + i];
    smem[OFF_Z + i] = zb[base_e + i];
  }
  for (int i = threadIdx.x; i < TCH * DS; i += 1024) {
    smem[OFF_B + i] = Bm[base_n + i];
    smem[OFF_C + i] = Cm[base_n + i];
  }
#endif
  __syncthreads();
  tdm_publish_fence(smem);              // buffer 0: DMA writes now visible

  int cur = 0;
  for (int c = 0; c < NCH; ++c) {
    const int nxt = cur ^ 1;
#if HAVE_TDM
    if (threadIdx.x == 0 && (c + 1) < NCH) {
      const long ge = base_e + (long)(c + 1) * TCH * DI;
      const long gn = base_n + (long)(c + 1) * TCH * DS;
      const unsigned bo = (unsigned)(nxt * BUF_F * 4);
      tdm_load_2d(delta + ge, bo + OFF_D * 4, DI, TCH);
      tdm_load_2d(xs    + ge, bo + OFF_X * 4, DI, TCH);
      tdm_load_2d(zb    + ge, bo + OFF_Z * 4, DI, TCH);
      tdm_load_2d(Bm    + gn, bo + OFF_B * 4, DS, TCH);
      tdm_load_2d(Cm    + gn, bo + OFF_C * 4, DS, TCH);
    }
#endif
    // volatile: the buffer was filled by the TDM engine, not by visible stores
    volatile const float* sb = smem + cur * BUF_F;
#pragma unroll 4
    for (int tt = 0; tt < TCH; ++tt) {
      const float dl = sb[OFF_D + tt * DI + e];
      const float xv = sb[OFF_X + tt * DI + e];
      const float Bv = sb[OFF_B + tt * DS + n];
      const float Cv = sb[OFF_C + tt * DS + n];
      const float dA = __expf(dl * A);
      h = dA * h + dl * Bv * xv;
      float p = h * Cv;
      p += __shfl_xor(p, 1, 16);
      p += __shfl_xor(p, 2, 16);
      p += __shfl_xor(p, 4, 16);
      p += __shfl_xor(p, 8, 16);
      if (n == 0) {
        const float zv = sb[OFF_Z + tt * DI + e];
        y[base_e + (long)(c * TCH + tt) * DI + e] = (p + Dk * xv) * silu_f(zv);
      }
    }
    __syncthreads();                // everyone done with buffer `cur`
#if HAVE_TDM
    if (threadIdx.x == 0 && (c + 1) < NCH)
      __builtin_amdgcn_s_wait_tensorcnt(0);   // chunk c+1 landed in LDS
#else
    if ((c + 1) < NCH) {
      const long ge = base_e + (long)(c + 1) * TCH * DI;
      const long gn = base_n + (long)(c + 1) * TCH * DS;
      float* sn = smem + nxt * BUF_F;
      for (int i = threadIdx.x; i < TCH * DI; i += 1024) {
        sn[OFF_D + i] = delta[ge + i];
        sn[OFF_X + i] = xs[ge + i];
        sn[OFF_Z + i] = zb[ge + i];
      }
      for (int i = threadIdx.x; i < TCH * DS; i += 1024) {
        sn[OFF_B + i] = Bm[gn + i];
        sn[OFF_C + i] = Cm[gn + i];
      }
    }
#endif
    __syncthreads();                // buffer `nxt` now valid for all waves
    tdm_publish_fence(smem);        // make the DMA-filled buffer visible
    cur = nxt;
  }
}

// ---------------------------------------------------------------------------
// Kernel 5a: mean over L of y -> ybar (B x 64)
// ---------------------------------------------------------------------------
__global__ void k_reduce(const float* __restrict__ y, float* __restrict__ ybar) {
  const int idx = blockIdx.x * blockDim.x + threadIdx.x;
  const int b = idx >> 6, e = idx & 63;
  float s = 0.f;
  for (int t = 0; t < L_; ++t) {
    if (t + 16 < L_) __builtin_prefetch(&y[((long)b * L_ + t + 16) * DI + e], 0, 0);
    s += y[((long)b * L_ + t) * DI + e];
  }
  ybar[idx] = s * (1.f / (float)L_);
}

// ---------------------------------------------------------------------------
// Kernel 5b: output GEMM folded into mean + static MLP + classifier head
// ---------------------------------------------------------------------------
__global__ void k_head(const float* __restrict__ ybar, const float* __restrict__ Wout,
                       const float* __restrict__ sf,  const float* __restrict__ Ws1,
                       const float* __restrict__ bs1, const float* __restrict__ Ws2,
                       const float* __restrict__ bs2, const float* __restrict__ Wc1,
                       const float* __restrict__ bc1, const float* __restrict__ Wc2,
                       const float* __restrict__ bc2, float* __restrict__ out) {
  const int b = threadIdx.x;
  if (b >= B_) return;
  float h1[32];
  for (int j = 0; j < 32; ++j) {
    float a = bs1[j];
    for (int i = 0; i < SD; ++i) a += sf[b * SD + i] * Ws1[i * 32 + j];
    h1[j] = a > 0.f ? a : 0.f;
  }
  float comb[DM];
  for (int j = 0; j < DM; ++j) {
    float a = bs2[j];
    for (int i = 0; i < 32; ++i) a += h1[i] * Ws2[i * DM + j];
    for (int e = 0; e < DI; ++e) a += ybar[b * DI + e] * Wout[e * DM + j];
    comb[j] = a;
  }
  float c1[16];
  for (int k = 0; k < 16; ++k) {
    float a = bc1[k];
    for (int j = 0; j < DM; ++j) a += comb[j] * Wc1[j * 16 + k];
    c1[k] = a > 0.f ? a : 0.f;
  }
  float lg = bc2[0];
  for (int k = 0; k < 16; ++k) lg += c1[k] * Wc2[k];
  out[b] = 1.f / (1.f + __expf(-lg));
}

// ---------------------------------------------------------------------------
extern "C" void kernel_launch(void* const* d_in, const int* in_sizes, int n_in,
                              void* d_out, int out_size, void* d_ws, size_t ws_size,
                              hipStream_t stream) {
  const float* tsv  = (const float*)d_in[0];
  const int*   ind  = (const int*)  d_in[1];
  const float* sf   = (const float*)d_in[2];
  const float* Wts  = (const float*)d_in[3];
  const float* bts  = (const float*)d_in[4];
  const float* Ws1  = (const float*)d_in[5];
  const float* bs1  = (const float*)d_in[6];
  const float* Ws2  = (const float*)d_in[7];
  const float* bs2  = (const float*)d_in[8];
  const float* Win  = (const float*)d_in[9];
  const float* cw   = (const float*)d_in[10];
  const float* cb   = (const float*)d_in[11];
  const float* Wx   = (const float*)d_in[12];
  const float* Wdt  = (const float*)d_in[13];
  const float* bdt  = (const float*)d_in[14];
  const float* Alog = (const float*)d_in[15];
  const float* Dsk  = (const float*)d_in[16];
  const float* Wout = (const float*)d_in[17];
  const float* Wc1  = (const float*)d_in[18];
  const float* bc1  = (const float*)d_in[19];
  const float* Wc2  = (const float*)d_in[20];
  const float* bc2  = (const float*)d_in[21];

  float* ws = (float*)d_ws;
  const long N_BL = (long)B_ * L_;
  float* tsp   = ws;                         // B*L*32
  float* xr    = tsp + N_BL * DM;            // B*L*64  (reused as y after scan)
  float* zb    = xr  + N_BL * DI;            // B*L*64
  float* xs    = zb  + N_BL * DI;            // B*L*64
  float* delta = xs  + N_BL * DI;            // B*L*64
  float* Bm    = delta + N_BL * DI;          // B*L*16
  float* Cm    = Bm  + N_BL * DS;            // B*L*16
  float* ybar  = Cm  + N_BL * DS;            // B*64
  float* y     = xr;                         // alias: x_raw dead after conv

  const int tiles = (int)(N_BL / 16);        // 4096 row-tiles
  k_maskproj<<<tiles / 4, 128, 0, stream>>>(tsv, ind, Wts, bts, tsp);
  k_inproj  <<<tiles / 4, 128, 0, stream>>>(tsp, Win, xr, zb);
  k_conv    <<<(int)(N_BL * DI / 256), 256, 0, stream>>>(xr, cw, cb, xs);
  k_dbc     <<<(int)(N_BL / 256), 256, 0, stream>>>(xs, Wx, Wdt, bdt, delta, Bm, Cm);
  k_scan    <<<B_, 1024, 0, stream>>>(xs, delta, Bm, Cm, zb, Alog, Dsk, y);
  k_reduce  <<<(B_ * DI) / 256, 256, 0, stream>>>(y, ybar);
  k_head    <<<1, 32, 0, stream>>>(ybar, Wout, sf, Ws1, bs1, Ws2, bs2,
                                   Wc1, bc1, Wc2, bc2, (float*)d_out);
}
